// Int8LlamaAttention_4861902979322
// MI455X (gfx1250) — 16-bit and 8-bit matrix operand order repaired and checked by reading, compiled, not run on hardware
//
#include <hip/hip_runtime.h>


typedef __attribute__((ext_vector_type(8)))  int      v8i;
typedef __attribute__((ext_vector_type(8)))  float    v8f;
typedef __attribute__((ext_vector_type(16))) _Float16 v16h;

#define B_    2
#define S_    1024
#define HID_  4096
#define NH_   32
#define HD_   128
#define A_SCALE_ 0.02f
#define W_SCALE_ 0.01f
#define O_SCALE_ 0.05f

union FragH  { v16h h; int4 q[2]; };
union FragA8 { v8i  v; int2 d[4]; };
union FragB8 { v8i  v; int4 d[2]; };

__device__ inline v8f v8f_zero() {
  v8f v;
#pragma unroll
  for (int i = 0; i < 8; ++i) v[i] = 0.f;
  return v;
}
__device__ inline v8i v8i_zero() {
  v8i v;
#pragma unroll
  for (int i = 0; i < 8; ++i) v[i] = 0;
  return v;
}

#if defined(__has_builtin)
#if __has_builtin(__builtin_amdgcn_global_load_async_to_lds_b128)
#define HAVE_ASYNC_LDS 1
#endif
#endif

typedef int vi4_ __attribute__((__vector_size__(16)));
typedef __attribute__((address_space(1))) vi4_ gvi4;
typedef __attribute__((address_space(3))) vi4_ lvi4;

__device__ inline gvi4* as_global4(const void* p) {
  return (gvi4*)(unsigned long long)p;
}
__device__ inline lvi4* as_lds4(void* p) {
  return (lvi4*)(unsigned int)(unsigned long long)p;
}

__device__ inline void stage32(const signed char* src, signed char* dst) {
#ifdef HAVE_ASYNC_LDS
  __builtin_amdgcn_global_load_async_to_lds_b128(as_global4(src), as_lds4(dst),
                                                 0, 0);
  __builtin_amdgcn_global_load_async_to_lds_b128(as_global4(src + 16),
                                                 as_lds4(dst + 16), 0, 0);
#else
  *(int4*)dst = *(const int4*)src;
  *(int4*)(dst + 16) = *(const int4*)(src + 16);
#endif
}

__device__ inline void wait_async0() {
#ifdef HAVE_ASYNC_LDS
#if __has_builtin(__builtin_amdgcn_s_wait_asynccnt)
  __builtin_amdgcn_s_wait_asynccnt(0);
#else
  asm volatile("s_wait_asynccnt 0" ::: "memory");
#endif
#endif
}

__global__ void __launch_bounds__(256) pack_w8(const int* __restrict__ w,
                                               signed char* __restrict__ out,
                                               int n4) {
  int i = blockIdx.x * blockDim.x + threadIdx.x;
  if (i < n4) {
    int4 v = ((const int4*)w)[i];
    ((char4*)out)[i] = make_char4((signed char)v.x, (signed char)v.y,
                                  (signed char)v.z, (signed char)v.w);
  }
}

__global__ void __launch_bounds__(256) quant_act(const float* __restrict__ x,
                                                 signed char* __restrict__ out,
                                                 int n4, float inv_scale) {
  int i = blockIdx.x * blockDim.x + threadIdx.x;
  if (i < n4) {
    float4 v = ((const float4*)x)[i];
    float a = fminf(fmaxf(rintf(v.x * inv_scale), -127.f), 127.f);
    float b = fminf(fmaxf(rintf(v.y * inv_scale), -127.f), 127.f);
    float c = fminf(fmaxf(rintf(v.z * inv_scale), -127.f), 127.f);
    float d = fminf(fmaxf(rintf(v.w * inv_scale), -127.f), 127.f);
    ((char4*)out)[i] = make_char4((signed char)a, (signed char)b,
                                  (signed char)c, (signed char)d);
  }
}

__device__ inline v8i load_a8(const signed char* __restrict__ Arow, int k0,
                              int sel) {
  FragA8 a;
  const int2* p = (const int2*)(Arow + k0);
  a.d[0] = p[sel];
  a.d[1] = p[2 + sel];
  a.d[2] = p[4 + sel];
  a.d[3] = p[6 + sel];
  return a.v;
}

__device__ inline v8i load_b8_lds(const signed char* __restrict__ ldsRow,
                                  int sel) {
  FragA8 b;
  const int2* p = (const int2*)ldsRow;
  b.d[0] = p[sel];
  b.d[1] = p[2 + sel];
  b.d[2] = p[4 + sel];
  b.d[3] = p[6 + sel];
  return b.v;
}

#define GEMM_STAGE (128 * 64)

__device__ inline void i8_gemm_mainloop(const signed char* __restrict__ Arow,
                                        const signed char* __restrict__ Wst,
                                        signed char* __restrict__ lds0,
                                        signed char* __restrict__ lds1,
                                        int stageOff, int nl, int sel,
                                        v8i acc[8]) {
  stage32(Wst, lds0 + stageOff);

  for (int kb = 0; kb < 64; ++kb) {
    const int k0 = kb * 64;
    wait_async0();
    __syncthreads();

    const int kn = (kb + 1) & 63;
    signed char* nbuf = ((kb & 1) ? lds0 : lds1) + stageOff;
    stage32(Wst + kn * 64, nbuf);

    __builtin_prefetch(Arow + k0 + 64, 0, 3);

    const signed char* cbuf = (kb & 1) ? lds1 : lds0;
    v8i a = load_a8(Arow, k0, sel);
#pragma unroll
    for (int t = 0; t < 8; ++t) {
      v8i b = load_b8_lds(cbuf + (t * 16 + nl) * 64, sel);
      acc[t] = __builtin_amdgcn_wmma_i32_16x16x64_iu8(true, a, true, b, acc[t],
                                                      false, false);
    }
  }
}

__global__ void __launch_bounds__(256) gemm_qkv(
    const signed char* __restrict__ xq, const signed char* __restrict__ w8base,
    const float* __restrict__ bq, const float* __restrict__ bk,
    const float* __restrict__ bv, float* __restrict__ qf,
    float* __restrict__ kf, _Float16* __restrict__ vT) {
  __shared__ signed char ldsW[2][GEMM_STAGE];

  const int z = blockIdx.z;
  const signed char* __restrict__ W = w8base + (size_t)z * HID_ * HID_;
  const float* __restrict__ bias = (z == 0) ? bq : (z == 1) ? bk : bv;

  const int lane = threadIdx.x & 31;
  const int wave = threadIdx.x >> 5;
  const int nl = lane & 15;
  const int sel = lane >> 4;
  const int colBase = blockIdx.y * 128;
  const int rowWave = blockIdx.x * 128 + wave * 16;
  const signed char* __restrict__ Arow = xq + (size_t)(rowWave + nl) * HID_;

  const int wr = threadIdx.x >> 1;
  const int wc = (threadIdx.x & 1) * 32;
  const signed char* __restrict__ Wst = W + (size_t)(colBase + wr) * HID_ + wc;
  const int stageOff = wr * 64 + wc;

  v8i acc[8];
#pragma unroll
  for (int t = 0; t < 8; ++t) acc[t] = v8i_zero();

  i8_gemm_mainloop(Arow, Wst, ldsW[0], ldsW[1], stageOff, nl, sel, acc);

  const float scale = A_SCALE_ * W_SCALE_;
#pragma unroll
  for (int t = 0; t < 8; ++t) {
    const int o = colBase + t * 16 + nl;
    const int h = o >> 7, d = o & 127;
    const float bo_ = bias[o];
#pragma unroll
    for (int j = 0; j < 8; ++j) {
      const int r = rowWave + j + 8 * sel;
      const int b = r >> 10, s = r & (S_ - 1);
      const float y = (float)acc[t][j] * scale + bo_;
      if (z == 2) {
        vT[(((size_t)(b * NH_ + h)) * HD_ + d) * S_ + s] = (_Float16)y;
      } else {
        float* dst = (z == 0) ? qf : kf;
        dst[(((size_t)(b * NH_ + h)) * S_ + s) * HD_ + d] = y;
      }
    }
  }
}

__global__ void __launch_bounds__(256) rope_kernel(
    const float* __restrict__ qf, const float* __restrict__ kf,
    _Float16* __restrict__ q16, _Float16* __restrict__ k16) {
  const int idx = blockIdx.x * blockDim.x + threadIdx.x;
  const int i = idx & 63;
  const int s = (idx >> 6) & (S_ - 1);
  const int bh = idx >> 16;
  const float inv = __expf(-0.1439115682f * (float)i);
  const float ang = (float)s * inv;
  float sn, cs;
  __sincosf(ang, &sn, &cs);
  const size_t base = ((size_t)bh * S_ + s) * HD_;
  const float q0 = qf[base + i], q1 = qf[base + i + 64];
  q16[base + i] = (_Float16)(q0 * cs - q1 * sn);
  q16[base + i + 64] = (_Float16)(q1 * cs + q0 * sn);
  const float k0 = kf[base + i], k1 = kf[base + i + 64];
  k16[base + i] = (_Float16)(k0 * cs - k1 * sn);
  k16[base + i + 64] = (_Float16)(k1 * cs + k0 * sn);
}

__global__ void __launch_bounds__(128) attn_kernel(
    const _Float16* __restrict__ q16, const _Float16* __restrict__ k16,
    const _Float16* __restrict__ vT, signed char* __restrict__ attn8) {
  __shared__ _Float16 pbuf[4 * 16 * 32];

  const int lane = threadIdx.x & 31;
  const int wave = threadIdx.x >> 5;
  const int nl = lane & 15;
  const int sel = lane >> 4;
  const int bh = blockIdx.y;
  const int bi = bh >> 5;
  const int hi = bh & 31;
  const int qBase = blockIdx.x * 64 + wave * 16;
  const size_t headQK = (size_t)bh * S_ * HD_;
  const size_t headV = (size_t)bh * HD_ * S_;

  FragH qa[4];
  {
    const int4* qp = (const int4*)(q16 + headQK + (size_t)(qBase + nl) * HD_);
#pragma unroll
    for (int c = 0; c < 4; ++c) {
      qa[c].q[0] = qp[c * 4 + sel];
      qa[c].q[1] = qp[c * 4 + 2 + sel];
    }
  }

  v8f acc[8];
#pragma unroll
  for (int f = 0; f < 8; ++f) acc[f] = v8f_zero();
  float mrow[8], lrow[8];
#pragma unroll
  for (int j = 0; j < 8; ++j) { mrow[j] = -1e30f; lrow[j] = 0.f; }

  const int nChunks = ((qBase + 15) >> 5) + 1;
  const float SC = 0.0883883476f;
  _Float16* __restrict__ pw = pbuf + wave * 512;

  for (int kb = 0; kb < nChunks; ++kb) {
    v8f s0 = v8f_zero(), s1 = v8f_zero();
#pragma unroll
    for (int t = 0; t < 2; ++t) {
      const int keyRow = kb * 32 + t * 16 + nl;
      const int4* kp = (const int4*)(k16 + headQK + (size_t)keyRow * HD_);
      v8f sc = v8f_zero();
#pragma unroll
      for (int c = 0; c < 4; ++c) {
        FragH kfr;
        kfr.q[0] = kp[c * 4 + sel];
        kfr.q[1] = kp[c * 4 + 2 + sel];
        sc = __builtin_amdgcn_wmma_f32_16x16x32_f16(false, qa[c].h, false,
                                                    kfr.h, (short)0, sc, false,
                                                    false);
      }
      if (t == 0) s0 = sc; else s1 = sc;
    }

#pragma unroll
    for (int j = 0; j < 8; ++j) {
      const int qrow = qBase + j + 8 * sel;
      const int key0 = kb * 32 + nl;
      const int key1 = key0 + 16;
      const bool ok0 = (key0 <= qrow), ok1 = (key1 <= qrow);
      const float a0 = ok0 ? s0[j] * SC : -1e30f;
      const float a1 = ok1 ? s1[j] * SC : -1e30f;
      float mx = fmaxf(a0, a1);
      mx = fmaxf(mx, __shfl_xor(mx, 1));
      mx = fmaxf(mx, __shfl_xor(mx, 2));
      mx = fmaxf(mx, __shfl_xor(mx, 4));
      mx = fmaxf(mx, __shfl_xor(mx, 8));
      const float nm = fmaxf(mrow[j], mx);
      const float alpha = __expf(mrow[j] - nm);
      mrow[j] = nm;
      const float p0 = ok0 ? __expf(a0 - nm) : 0.f;
      const float p1 = ok1 ? __expf(a1 - nm) : 0.f;
      float ps = p0 + p1;
      ps += __shfl_xor(ps, 1);
      ps += __shfl_xor(ps, 2);
      ps += __shfl_xor(ps, 4);
      ps += __shfl_xor(ps, 8);
      lrow[j] = lrow[j] * alpha + ps;
#pragma unroll
      for (int f = 0; f < 8; ++f) acc[f][j] *= alpha;
      pw[(j + 8 * sel) * 32 + nl] = (_Float16)p0;
      pw[(j + 8 * sel) * 32 + 16 + nl] = (_Float16)p1;
    }

    FragH pa;
    {
      const int4* pp = (const int4*)(pw + nl * 32);
      pa.q[0] = pp[sel];
      pa.q[1] = pp[2 + sel];
    }

#pragma unroll
    for (int f = 0; f < 8; ++f) {
      const int4* vp =
          (const int4*)(vT + headV + (size_t)(f * 16 + nl) * S_ + kb * 32);
      FragH vb;
      vb.q[0] = vp[sel];
      vb.q[1] = vp[2 + sel];
      acc[f] = __builtin_amdgcn_wmma_f32_16x16x32_f16(false, pa.h, false, vb.h,
                                                      (short)0, acc[f], false,
                                                      false);
    }
  }

  const float invO = 1.f / O_SCALE_;
#pragma unroll
  for (int j = 0; j < 8; ++j) {
    const int qrow = qBase + j + 8 * sel;
    const float invl = 1.f / lrow[j];
    signed char* dst =
        attn8 + ((size_t)bi * S_ + qrow) * HID_ + hi * HD_ + nl;
#pragma unroll
    for (int f = 0; f < 8; ++f) {
      const float y = acc[f][j] * invl;
      const float qv = fminf(fmaxf(rintf(y * invO), -127.f), 127.f);
      dst[f * 16] = (signed char)qv;
    }
  }
}

__global__ void __launch_bounds__(256) gemm_o(const signed char* __restrict__ a8,
                                              const signed char* __restrict__ wo8,
                                              const float* __restrict__ bo,
                                              float* __restrict__ out) {
  __shared__ signed char ldsW[2][GEMM_STAGE];

  const int lane = threadIdx.x & 31;
  const int wave = threadIdx.x >> 5;
  const int nl = lane & 15;
  const int sel = lane >> 4;
  const int colBase = blockIdx.y * 128;
  const int rowWave = blockIdx.x * 128 + wave * 16;
  const signed char* __restrict__ Arow = a8 + (size_t)(rowWave + nl) * HID_;

  const int wr = threadIdx.x >> 1;
  const int wc = (threadIdx.x & 1) * 32;
  const signed char* __restrict__ Wst =
      wo8 + (size_t)(colBase + wr) * HID_ + wc;
  const int stageOff = wr * 64 + wc;

  v8i acc[8];
#pragma unroll
  for (int t = 0; t < 8; ++t) acc[t] = v8i_zero();

  i8_gemm_mainloop(Arow, Wst, ldsW[0], ldsW[1], stageOff, nl, sel, acc);

  const float scale = O_SCALE_ * W_SCALE_;
#pragma unroll
  for (int t = 0; t < 8; ++t) {
    const int o = colBase + t * 16 + nl;
    const float bo_ = bo[o];
#pragma unroll
    for (int j = 0; j < 8; ++j) {
      const int r = rowWave + j + 8 * sel;
      out[(size_t)r * HID_ + o] = (float)acc[t][j] * scale + bo_;
    }
  }
}

extern "C" void kernel_launch(void* const* d_in, const int* in_sizes, int n_in,
                              void* d_out, int out_size, void* d_ws,
                              size_t ws_size, hipStream_t stream) {
  (void)in_sizes; (void)n_in; (void)out_size; (void)ws_size;

  const float* hs = (const float*)d_in[0];
  const int* wq = (const int*)d_in[3];
  const int* wk = (const int*)d_in[4];
  const int* wv = (const int*)d_in[5];
  const int* wo = (const int*)d_in[6];
  const float* bq = (const float*)d_in[7];
  const float* bk = (const float*)d_in[8];
  const float* bv = (const float*)d_in[9];
  const float* bo = (const float*)d_in[10];

  const size_t SLAB = (size_t)HID_ * HID_;
  char* ws = (char*)d_ws;
  signed char* w8 = (signed char*)(ws);
  signed char* xq = (signed char*)(ws + 4 * SLAB);
  float* qf = (float*)(ws + 4 * SLAB + 8388608);
  float* kf = (float*)(ws + 4 * SLAB + 8388608 + 33554432);
  _Float16* vT = (_Float16*)(ws + 4 * SLAB + 8388608 + 2 * 33554432ull);
  _Float16* q16 =
      (_Float16*)(ws + 4 * SLAB + 8388608 + 2 * 33554432ull + 16777216);
  _Float16* k16 =
      (_Float16*)(ws + 4 * SLAB + 8388608 + 2 * 33554432ull + 2 * 16777216ull);
  signed char* a8 =
      (signed char*)(ws + 4 * SLAB + 8388608 + 2 * 33554432ull +
                     3 * 16777216ull);

  const int n4w = (int)(SLAB / 4);
  pack_w8<<<n4w / 256, 256, 0, stream>>>(wq, w8 + 0 * SLAB, n4w);
  pack_w8<<<n4w / 256, 256, 0, stream>>>(wk, w8 + 1 * SLAB, n4w);
  pack_w8<<<n4w / 256, 256, 0, stream>>>(wv, w8 + 2 * SLAB, n4w);
  pack_w8<<<n4w / 256, 256, 0, stream>>>(wo, w8 + 3 * SLAB, n4w);

  const int n4a = B_ * S_ * HID_ / 4;
  quant_act<<<n4a / 256, 256, 0, stream>>>(hs, xq, n4a, 1.f / A_SCALE_);

  dim3 gq(B_ * S_ / 128, HID_ / 128, 3);
  gemm_qkv<<<gq, 256, 0, stream>>>(xq, w8, bq, bk, bv, qf, kf, vT);

  const int nrope = B_ * NH_ * S_ * (HD_ / 2);
  rope_kernel<<<nrope / 256, 256, 0, stream>>>(qf, kf, q16, k16);

  dim3 ga(S_ / 64, B_ * NH_);
  attn_kernel<<<ga, 128, 0, stream>>>(q16, k16, vT, a8);

  dim3 go(B_ * S_ / 128, HID_ / 128);
  gemm_o<<<go, 256, 0, stream>>>(a8, w8 + 3 * SLAB, bo, (float*)d_out);
}
